// Head_13365938225281
// MI455X (gfx1250) — matrix operand order repaired and checked by reading, not compiled, not run on hardware
//
#include <hip/hip_runtime.h>
#include <hip/hip_bf16.h>
#include <hip/hip_fp16.h>

typedef __attribute__((ext_vector_type(16))) _Float16 v16h;
typedef __attribute__((ext_vector_type(8)))  _Float16 v8h;
typedef __attribute__((ext_vector_type(8)))  float    v8f;

#define WMMA_F32_F16(a, b, c) \
  __builtin_amdgcn_wmma_f32_16x16x32_f16(false, (a), false, (b), (short)0, (c), false, false)

static __device__ __forceinline__ v16h cat8(v8h lo, v8h hi) {
  return __builtin_shufflevector(lo, hi, 0,1,2,3,4,5,6,7,8,9,10,11,12,13,14,15);
}
static __device__ __forceinline__ v8f vzero8() {
  v8f r;
#pragma unroll
  for (int i = 0; i < 8; ++i) r[i] = 0.0f;
  return r;
}

// ---------------------------------------------------------------------------
// Kernel 0: convert W[1024][64] f32 -> Wt[m][64][1024] f16 (transposed), m=K,Q,V
// ---------------------------------------------------------------------------
__global__ void prep_weights(const float* __restrict__ Wk,
                             const float* __restrict__ Wq,
                             const float* __restrict__ Wv,
                             _Float16* __restrict__ Wt) {
  int idx = blockIdx.x * blockDim.x + threadIdx.x;   // 0 .. 3*64*1024-1
  if (idx >= 3 * 64 * 1024) return;
  int m    = idx >> 16;         // 65536 per matrix
  int rem  = idx & 65535;
  int ncol = rem >> 10;         // 0..63
  int k    = rem & 1023;        // 0..1023
  const float* W = (m == 0) ? Wk : (m == 1) ? Wq : Wv;
  Wt[idx] = (_Float16)W[k * 64 + ncol];
}

// ---------------------------------------------------------------------------
// Kernel 1: projections. One wave per (16-row tile, matrix) job.
//   K, Q stored f16 row-major [8192][64]; V stored transposed [4][64][2048].
// ---------------------------------------------------------------------------
__global__ __launch_bounds__(256) void proj_kernel(const float* __restrict__ x,
                                                   const _Float16* __restrict__ Wt,
                                                   _Float16* __restrict__ Kh,
                                                   _Float16* __restrict__ Qh,
                                                   _Float16* __restrict__ Vt) {
  const int lane = threadIdx.x & 31;
  const int wv   = threadIdx.x >> 5;
  const int job  = blockIdx.x * 8 + wv;   // 0..1535
  const int rt   = job / 3;               // row tile 0..511
  const int m    = job - rt * 3;          // 0=K 1=Q 2=V
  const int n16  = lane & 15;
  const int h    = lane >> 4;

  const float*    xrow = x + (size_t)(rt * 16 + n16) * 1024;   // A row = lane%16
  const _Float16* Wm   = Wt + m * 65536;

  v8f acc[4];
#pragma unroll
  for (int nt = 0; nt < 4; ++nt) acc[nt] = vzero8();

  for (int k0 = 0; k0 < 1024; k0 += 32) {
    // A tile (16x32 f16): lane holds row n16; halves [8h..8h+7] and [16+8h..16+8h+7]
    const float4 f0 = *(const float4*)(xrow + k0 + 8 * h);
    const float4 f1 = *(const float4*)(xrow + k0 + 8 * h + 4);
    const float4 f2 = *(const float4*)(xrow + k0 + 16 + 8 * h);
    const float4 f3 = *(const float4*)(xrow + k0 + 16 + 8 * h + 4);
    v16h a;
    a[0]=(_Float16)f0.x; a[1]=(_Float16)f0.y; a[2]=(_Float16)f0.z; a[3]=(_Float16)f0.w;
    a[4]=(_Float16)f1.x; a[5]=(_Float16)f1.y; a[6]=(_Float16)f1.z; a[7]=(_Float16)f1.w;
    a[8]=(_Float16)f2.x; a[9]=(_Float16)f2.y; a[10]=(_Float16)f2.z; a[11]=(_Float16)f2.w;
    a[12]=(_Float16)f3.x; a[13]=(_Float16)f3.y; a[14]=(_Float16)f3.z; a[15]=(_Float16)f3.w;
#pragma unroll
    for (int nt = 0; nt < 4; ++nt) {
      // B tile (32x16): lane holds col n16; halves [8h..8h+7] and [16+8h..16+8h+7]
      const _Float16* wrow = Wm + (nt * 16 + n16) * 1024 + k0 + 8 * h;
      v16h b = cat8(*(const v8h*)(wrow), *(const v8h*)(wrow + 16));
      acc[nt] = WMMA_F32_F16(a, b, acc[nt]);
    }
  }

#pragma unroll
  for (int nt = 0; nt < 4; ++nt) {
#pragma unroll
    for (int j = 0; j < 8; ++j) {
      const int row = rt * 16 + 8 * h + j;       // C/D layout: row = 8h + vgpr
      const int col = nt * 16 + n16;
      const _Float16 val = (_Float16)acc[nt][j];
      if (m == 0)      Kh[row * 64 + col] = val;
      else if (m == 1) Qh[row * 64 + col] = val;
      else {
        const int b2 = row >> 11, t = row & 2047;
        Vt[b2 * 131072 + col * 2048 + t] = val;  // V transposed per batch
      }
    }
  }
}

// ---------------------------------------------------------------------------
// Kernel 2: flash-style causal attention. One wave per 16-query tile.
// ---------------------------------------------------------------------------
__global__ __launch_bounds__(256) void attn_kernel(const _Float16* __restrict__ Qh,
                                                   const _Float16* __restrict__ Kh,
                                                   const _Float16* __restrict__ Vt,
                                                   float* __restrict__ out) {
  __shared__ _Float16 Pbuf[8][16 * 32];          // per-wave P scratch (1 KB each)
  const int lane = threadIdx.x & 31;
  const int wv   = threadIdx.x >> 5;
  const int job  = blockIdx.x * 8 + wv;          // 0..511
  const int b    = job >> 7;                     // batch
  const int qt   = job & 127;                    // query tile in batch
  const int q_base = qt * 16;
  const int n16 = lane & 15;
  const int h   = lane >> 4;

  const _Float16* Qb = Qh + (size_t)b * 131072;
  const _Float16* Kb = Kh + (size_t)b * 131072;
  const _Float16* Vb = Vt + (size_t)b * 131072;

  // Q as A-matrix, two K-steps over head dim 64
  const _Float16* qrow = Qb + (q_base + n16) * 64;
  const v16h aq0 = cat8(*(const v8h*)(qrow + 8 * h),      *(const v8h*)(qrow + 16 + 8 * h));
  const v16h aq1 = cat8(*(const v8h*)(qrow + 32 + 8 * h), *(const v8h*)(qrow + 48 + 8 * h));

  v8f o[4];
#pragma unroll
  for (int nt = 0; nt < 4; ++nt) o[nt] = vzero8();
  float mrun[8], lrun[8];
#pragma unroll
  for (int j = 0; j < 8; ++j) { mrun[j] = -3.0e38f; lrun[j] = 0.0f; }

  _Float16* P = &Pbuf[wv][0];
  const int n_kb = ((q_base + 15) >> 5) + 1;     // 32-key blocks touching diagonal

  for (int kb = 0; kb < n_kb; ++kb) {
    const int s_base = kb * 32;

    // ---- S = Q K^T for two 16-key tiles ----
    v8f s0 = vzero8(), s1 = vzero8();
    {
      const _Float16* krow = Kb + (s_base + n16) * 64;
      v16h bk0 = cat8(*(const v8h*)(krow + 8 * h),      *(const v8h*)(krow + 16 + 8 * h));
      v16h bk1 = cat8(*(const v8h*)(krow + 32 + 8 * h), *(const v8h*)(krow + 48 + 8 * h));
      s0 = WMMA_F32_F16(aq0, bk0, s0);
      s0 = WMMA_F32_F16(aq1, bk1, s0);
      const _Float16* krow2 = krow + 16 * 64;
      v16h bk2 = cat8(*(const v8h*)(krow2 + 8 * h),      *(const v8h*)(krow2 + 16 + 8 * h));
      v16h bk3 = cat8(*(const v8h*)(krow2 + 32 + 8 * h), *(const v8h*)(krow2 + 48 + 8 * h));
      s1 = WMMA_F32_F16(aq0, bk2, s1);
      s1 = WMMA_F32_F16(aq1, bk3, s1);
    }

    // ---- online softmax in fp32 (C/D layout: row = 8h+j, col = n16) ----
    float cj[8];
#pragma unroll
    for (int j = 0; j < 8; ++j) {
      const int q = q_base + 8 * h + j;
      float a0 = s0[j] * 0.03125f;               // * C^-0.5 = 1/32
      float a1 = s1[j] * 0.03125f;
      if (s_base + n16 > q)      a0 = -1.0e30f;  // causal mask
      if (s_base + 16 + n16 > q) a1 = -1.0e30f;
      float bm = fmaxf(a0, a1);
#pragma unroll
      for (int msk = 8; msk > 0; msk >>= 1) bm = fmaxf(bm, __shfl_xor(bm, msk, 32));
      const float mn = fmaxf(mrun[j], bm);
      const float c  = __expf(mrun[j] - mn);
      const float p0 = __expf(a0 - mn);
      const float p1 = __expf(a1 - mn);
      float rs = p0 + p1;
#pragma unroll
      for (int msk = 8; msk > 0; msk >>= 1) rs += __shfl_xor(rs, msk, 32);
      lrun[j] = lrun[j] * c + rs;
      mrun[j] = mn;
      cj[j] = c;
      P[(8 * h + j) * 32 + n16]      = (_Float16)p0;   // P row-major 16x32 f16
      P[(8 * h + j) * 32 + 16 + n16] = (_Float16)p1;
    }
#pragma unroll
    for (int nt = 0; nt < 4; ++nt)
#pragma unroll
      for (int j = 0; j < 8; ++j) o[nt][j] *= cj[j];

    // ---- re-layout P (C-layout) -> A-layout via LDS; DS ops are in-order ----
    asm volatile("s_wait_dscnt 0x0" ::: "memory");
    const v16h ap = cat8(*(const v8h*)(P + n16 * 32 + 8 * h),
                         *(const v8h*)(P + n16 * 32 + 16 + 8 * h));
    asm volatile("" ::: "memory");

    // ---- O += P * V (V transposed: contiguous keys per head-col) ----
#pragma unroll
    for (int nt = 0; nt < 4; ++nt) {
      const _Float16* vcol = Vb + (nt * 16 + n16) * 2048 + s_base + 8 * h;
      v16h bv = cat8(*(const v8h*)(vcol), *(const v8h*)(vcol + 16));
      o[nt] = WMMA_F32_F16(ap, bv, o[nt]);
    }
  }

  // ---- normalize and store fp32 output ----
  float* obase = out + (size_t)(b * 2048 + q_base) * 64;
#pragma unroll
  for (int nt = 0; nt < 4; ++nt)
#pragma unroll
    for (int j = 0; j < 8; ++j)
      obase[(8 * h + j) * 64 + nt * 16 + n16] = o[nt][j] / lrun[j];
}

// ---------------------------------------------------------------------------
extern "C" void kernel_launch(void* const* d_in, const int* in_sizes, int n_in,
                              void* d_out, int out_size, void* d_ws, size_t ws_size,
                              hipStream_t stream) {
  const float* x  = (const float*)d_in[0];
  const float* Wk = (const float*)d_in[1];
  const float* Wq = (const float*)d_in[2];
  const float* Wv = (const float*)d_in[3];
  float* out = (float*)d_out;

  char* ws = (char*)d_ws;
  _Float16* Wt = (_Float16*)(ws);                          // 3*64*1024 f16 = 384 KB
  _Float16* Kh = (_Float16*)(ws + 393216);                 // 8192*64 f16 = 1 MB
  _Float16* Qh = (_Float16*)(ws + 393216 + 1048576);       // 1 MB
  _Float16* Vt = (_Float16*)(ws + 393216 + 2097152);       // 1 MB (transposed)

  prep_weights<<<768, 256, 0, stream>>>(Wk, Wq, Wv, Wt);   // 768*256 = 196608 threads
  proj_kernel<<<192, 256, 0, stream>>>(x, Wt, Kh, Qh, Vt); // 1536 wave-jobs
  attn_kernel<<<64, 256, 0, stream>>>(Qh, Kh, Vt, out);    // 512 wave-jobs
}
